// FocallyModulatedSupConLoss_61907658604687
// MI455X (gfx1250) — matrix operand order repaired and checked by reading, not compiled, not run on hardware
//
#include <hip/hip_runtime.h>
#include <hip/hip_bf16.h>

typedef __attribute__((ext_vector_type(16))) _Float16 v16h;
typedef __attribute__((ext_vector_type(8)))  _Float16 v8h;
typedef __attribute__((ext_vector_type(8)))  float    v8f;

// Problem constants (from reference setup_inputs)
constexpr int   Bn    = 2048;
constexpr int   Vv    = 2;
constexpr int   Dd    = 256;
constexpr int   Cc    = 1000;
constexpr int   Nn    = Vv * Bn;          // 4096
constexpr float INV_T = 1.0f / 0.07f;     // TEMPERATURE; T/BASE_T == 1
constexpr int   ROWS_PER_WG = 128;        // 8 waves * 16 rows
constexpr int   COLS_PER_WG = 512;        // grid.y = Nn/512 = 8 column blocks
constexpr int   NCB   = Nn / COLS_PER_WG; // 8
constexpr int   KCH   = Dd / 32;          // 8 wmma k-chunks
constexpr int   LDSTR = 24;               // LDS row stride in halves (bank-tuned)

// ---------------------------------------------------------------- init
__global__ __launch_bounds__(256) void init_kernel(int* count) {
  int i = blockIdx.x * 256 + threadIdx.x;
  if (i < Cc) count[i] = 0;
}

// --------------------------------------------------------- label histogram
__global__ __launch_bounds__(256) void hist_kernel(const int* __restrict__ labels,
                                                   int* __restrict__ count) {
  int i = blockIdx.x * 256 + threadIdx.x;
  if (i < Bn) atomicAdd(&count[labels[i]], 1);
}

// ------------------------------------------------- normalize + f16 copies
// one wave per row i of cf: cf[i] = features[b][v] / ||.||, i = v*B + b
// cf  : row-major [N][D]          (A fragments)
// cfP : panel-major [N/16][D][16] (B tiles; contiguous 8KB per 16-col panel)
__global__ __launch_bounds__(256) void normalize_kernel(
    const float* __restrict__ feat, _Float16* __restrict__ cf,
    _Float16* __restrict__ cfP) {
  int gw   = (blockIdx.x * 256 + threadIdx.x) >> 5;   // global wave = row i
  int lane = threadIdx.x & 31;
  if (gw >= Nn) return;
  int v = gw >> 11, b = gw & (Bn - 1);                // i = v*B + b
  const float* src = feat + ((size_t)b * Vv + v) * Dd;

  float x[8]; float ss = 0.0f;
#pragma unroll
  for (int e = 0; e < 8; ++e) { x[e] = src[lane * 8 + e]; ss += x[e] * x[e]; }
#pragma unroll
  for (int m = 1; m < 32; m <<= 1) ss += __shfl_xor(ss, m, 32);
  float rn = 1.0f / sqrtf(ss);

  v8h yv;
#pragma unroll
  for (int e = 0; e < 8; ++e) yv[e] = (_Float16)(x[e] * rn);
  *(v8h*)(cf + (size_t)gw * Dd + lane * 8) = yv;      // row-major copy
  _Float16* pbase = cfP + (size_t)(gw >> 4) * (Dd * 16) + (gw & 15);
#pragma unroll
  for (int e = 0; e < 8; ++e)                         // panel-major copy
    pbase[(size_t)(lane * 8 + e) * 16] = yv[e];
}

// ---------------------------------------------- focal modulation (1-pt)^2
__global__ __launch_bounds__(256) void modulation_kernel(
    const float* __restrict__ preds, const int* __restrict__ labels,
    float* __restrict__ mod) {
  int b = blockIdx.x;
  const float* row = preds + (size_t)b * Cc;
  __shared__ float red[8];
  int tid = threadIdx.x, lane = tid & 31, w = tid >> 5;

  float mx = -3.0e30f;
  for (int c = tid; c < Cc; c += 256) mx = fmaxf(mx, row[c]);
#pragma unroll
  for (int m = 1; m < 32; m <<= 1) mx = fmaxf(mx, __shfl_xor(mx, m, 32));
  if (lane == 0) red[w] = mx;
  __syncthreads();
  float bmx = red[0];
#pragma unroll
  for (int k = 1; k < 8; ++k) bmx = fmaxf(bmx, red[k]);
  __syncthreads();

  float se = 0.0f;
  for (int c = tid; c < Cc; c += 256) se += __expf(row[c] - bmx);
#pragma unroll
  for (int m = 1; m < 32; m <<= 1) se += __shfl_xor(se, m, 32);
  if (lane == 0) red[w] = se;
  __syncthreads();
  if (tid == 0) {
    float S = red[0] + red[1] + red[2] + red[3] + red[4] + red[5] + red[6] + red[7];
    float nll = row[labels[b]] - bmx - __logf(S);     // log p_t
    float pt  = __expf(nll);
    float m1  = 1.0f - pt;
    mod[b]    = m1 * m1;                              // GAMMA = 2
  }
}

// -------------------------------- fused WMMA Gram + per-row partial stats
// grid = (Nn/128, NCB); block = 256 (8 waves). Wave w owns rows row0+w*16;
// block.y owns a 512-column slice. Partials stored per (colblock,row):
//   sPart = sum_j exp(raw_ij - 1/T)   (ALL j incl diagonal)
//   pPart = sum_{label match incl diag} raw_ij
// Diagonal terms and counts are fixed up in finalize (no atomics anywhere).
__global__ __launch_bounds__(256) void gram_stats_kernel(
    const _Float16* __restrict__ cf, const _Float16* __restrict__ cfP,
    const int* __restrict__ labels, float* __restrict__ sPart,
    float* __restrict__ pPart) {
  __shared__ _Float16 ldsB[Dd * LDSTR];               // 12 KB staged B tile
  __shared__ int      ldsLab[COLS_PER_WG];            // 2 KB column labels
  int tid  = threadIdx.x;
  int lane = tid & 31;
  int wave = tid >> 5;
  int h    = lane >> 4;                               // half-wave selector
  int n16  = lane & 15;
  int row0 = blockIdx.x * ROWS_PER_WG + wave * 16;
  int col0 = blockIdx.y * COLS_PER_WG;

  // stage column labels once (first loop barrier publishes them)
  ldsLab[tid]       = labels[(col0 + tid) & (Bn - 1)];
  ldsLab[tid + 256] = labels[(col0 + tid + 256) & (Bn - 1)];

  // A fragments: this wave's 16-row strip, all K, in registers.
  v16h afrag[KCH];
  {
    const _Float16* abase = cf + (size_t)(row0 + n16) * Dd + h * 8;
#pragma unroll
    for (int c = 0; c < KCH; ++c) {
      v8h lo = *(const v8h*)(abase + c * 32);
      v8h hi = *(const v8h*)(abase + c * 32 + 16);
      v16h a;
#pragma unroll
      for (int e = 0; e < 8; ++e) { a[e] = lo[e]; a[8 + e] = hi[e]; }
      afrag[c] = a;
    }
  }
  int labRow[8];
#pragma unroll
  for (int r = 0; r < 8; ++r) labRow[r] = labels[(row0 + r + 8 * h) & (Bn - 1)];

  float sAc[8], pAc[8];
#pragma unroll
  for (int r = 0; r < 8; ++r) { sAc[r] = 0.0f; pAc[r] = 0.0f; }

  for (int jt = col0; jt < col0 + COLS_PER_WG; jt += 16) {
    __syncthreads();                                  // LDS reuse fence
    {                                                 // coalesced 8KB panel copy
      const _Float16* src = cfP + (size_t)(jt >> 4) * (Dd * 16) + tid * 16;
      __builtin_prefetch(src + Dd * 16, 0, 3);        // next panel
      v8h s0 = *(const v8h*)(src);
      v8h s1 = *(const v8h*)(src + 8);
      *(v8h*)(ldsB + tid * LDSTR)     = s0;
      *(v8h*)(ldsB + tid * LDSTR + 8) = s1;
    }
    __syncthreads();

    int labCol = ldsLab[(jt - col0) + n16];           // early ds_load_b32

    // preload ALL 8 B fragments, then chain WMMAs (pipelined dscnt waits)
    v16h bfrag[KCH];
#pragma unroll
    for (int c = 0; c < KCH; ++c) {
      const _Float16* bsrc = ldsB + (c * 32 + h * 8) * LDSTR + n16;
      v16h bf;
#pragma unroll
      for (int e = 0; e < 8; ++e) { bf[e] = bsrc[e * LDSTR]; bf[8 + e] = bsrc[(16 + e) * LDSTR]; }
      bfrag[c] = bf;
    }
    v8f acc = {};
#pragma unroll
    for (int c = 0; c < KCH; ++c)
      acc = __builtin_amdgcn_wmma_f32_16x16x32_f16(
          false, afrag[c], false, bfrag[c], (short)0, acc, false, false);

#pragma unroll
    for (int r = 0; r < 8; ++r) {                     // C elem r -> row r+8h
      float raw = acc[r] * INV_T;
      float e   = __expf(raw - INV_T);                // fixed shift M0 = 1/T
      sAc[r] += e;                                    // diag fixed in finalize
      pAc[r] += (labRow[r] == labCol) ? raw : 0.0f;
    }
  }

  // reduce the 16 lanes of each half-wave (distinct columns), store partials
  int cbOff = blockIdx.y * Nn;
#pragma unroll
  for (int r = 0; r < 8; ++r) {
    float s = sAc[r], p = pAc[r];
#pragma unroll
    for (int m = 1; m < 16; m <<= 1) {
      s += __shfl_xor(s, m, 16);
      p += __shfl_xor(p, m, 16);
    }
    if (n16 == 0) {
      int row = row0 + r + 8 * h;
      sPart[cbOff + row] = s;
      pPart[cbOff + row] = p;
    }
  }
}

// ------------------------------------------- per-row loss + block partials
__global__ __launch_bounds__(256) void finalize_kernel(
    const float* __restrict__ sPart, const float* __restrict__ pPart,
    const int* __restrict__ count, const int* __restrict__ labels,
    const float* __restrict__ mod, const _Float16* __restrict__ cf,
    float* __restrict__ bsum) {
  int i = blockIdx.x * 256 + threadIdx.x;             // row, i < Nn
  float s = 0.0f, p = 0.0f;
#pragma unroll
  for (int cb = 0; cb < NCB; ++cb) {                  // fixed-order merge
    s += sPart[cb * Nn + i];
    p += pPart[cb * Nn + i];
  }
  // recompute diagonal term raw_ii from the f16 row
  const v8h* crow = (const v8h*)(cf + (size_t)i * Dd);
  float ssq = 0.0f;
#pragma unroll
  for (int c = 0; c < Dd / 8; ++c) {
    v8h x = crow[c];
#pragma unroll
    for (int e = 0; e < 8; ++e) { float f = (float)x[e]; ssq += f * f; }
  }
  float rawii = ssq * INV_T;
  s -= __expf(rawii - INV_T);                         // drop self-contrast
  p -= rawii;
  int   lab  = labels[i & (Bn - 1)];
  float cnt  = (float)(2 * count[lab] - 1);           // analytic mask count
  float mlpp = p / cnt - (INV_T + __logf(s));
  float li   = -mod[i & (Bn - 1)] * mlpp;             // T/BASE_T == 1

  __shared__ float red[8];
  int lane = threadIdx.x & 31, w = threadIdx.x >> 5;
#pragma unroll
  for (int m = 1; m < 32; m <<= 1) li += __shfl_xor(li, m, 32);
  if (lane == 0) red[w] = li;
  __syncthreads();
  if (threadIdx.x == 0) {
    bsum[blockIdx.x] = red[0] + red[1] + red[2] + red[3] +
                       red[4] + red[5] + red[6] + red[7];
  }
}

// -------------------------------------------------- deterministic reduce
__global__ __launch_bounds__(32) void reduce_kernel(const float* __restrict__ bsum,
                                                    float* __restrict__ out) {
  int lane = threadIdx.x;
  float v = (lane < Nn / 256) ? bsum[lane] : 0.0f;
#pragma unroll
  for (int m = 1; m < 32; m <<= 1) v += __shfl_xor(v, m, 32);
  if (lane == 0) out[0] = v * (1.0f / (float)Nn);
}

// ---------------------------------------------------------------- host
extern "C" void kernel_launch(void* const* d_in, const int* in_sizes, int n_in,
                              void* d_out, int out_size, void* d_ws, size_t ws_size,
                              hipStream_t stream) {
  const float* feat   = (const float*)d_in[0];   // [B,V,D] f32
  const float* preds  = (const float*)d_in[1];   // [B,C]   f32
  const int*   labels = (const int*)d_in[2];     // [B]     i32
  float*       out    = (float*)d_out;           // scalar f32

  char* ws = (char*)d_ws;
  _Float16* cf    = (_Float16*)ws;                         // 2 MB
  _Float16* cfP   = cf + (size_t)Nn * Dd;                  // 2 MB
  float*    mod   = (float*)(ws + 2 * (size_t)Nn * Dd * sizeof(_Float16));
  float*    sPart = mod + Bn;                              // NCB*Nn
  float*    pPart = sPart + NCB * Nn;                      // NCB*Nn
  float*    bsum  = pPart + NCB * Nn;                      // 16
  int*      count = (int*)(bsum + Nn / 256);               // Cc

  init_kernel<<<(Cc + 255) / 256, 256, 0, stream>>>(count);
  hist_kernel<<<Bn / 256, 256, 0, stream>>>(labels, count);
  normalize_kernel<<<Nn / 8, 256, 0, stream>>>(feat, cf, cfP);
  modulation_kernel<<<Bn, 256, 0, stream>>>(preds, labels, mod);
  gram_stats_kernel<<<dim3(Nn / ROWS_PER_WG, NCB), 256, 0, stream>>>(
      cf, cfP, labels, sPart, pPart);
  finalize_kernel<<<Nn / 256, 256, 0, stream>>>(sPart, pPart, count, labels,
                                                mod, cf, bsum);
  reduce_kernel<<<1, 32, 0, stream>>>(bsum, out);
}
